// Net_33071248179767
// MI455X (gfx1250) — matrix operand order repaired and checked by reading, not compiled, not run on hardware
//
#include <hip/hip_runtime.h>
#include <hip/hip_bf16.h>

// ---------------- problem constants (from reference) ----------------
#define N_NODES 100000
#define E_EDGES 1600000
#define D_IN    100
#define K1PAD   128      // D_IN padded to multiple of 32 for bf16 WMMA
#define D_H     256
#define D_OUT   128
#define E_TOT   (E_EDGES + N_NODES)   // edges + self loops

typedef __bf16 bf16;
typedef __attribute__((ext_vector_type(16))) __bf16 bf16x16;
typedef __attribute__((ext_vector_type(8)))  __bf16 bf16x8;
typedef __attribute__((ext_vector_type(8)))  float  floatx8;

// ---------------- gcn_norm pieces ----------------
__global__ void k_init_deg(float* deg) {
    int i = blockIdx.x * blockDim.x + threadIdx.x;
    if (i < N_NODES) deg[i] = 1.0f;            // self-loop weight 1 pre-added
}

__global__ void k_deg_edges(const long long* __restrict__ col,
                            const float* __restrict__ w,
                            float* __restrict__ deg) {
    int e = blockIdx.x * blockDim.x + threadIdx.x;
    if (e < E_EDGES) atomicAdd(&deg[(int)col[e]], w[e]);
}

__global__ void k_dinv(float* deg) {
    int i = blockIdx.x * blockDim.x + threadIdx.x;
    if (i < N_NODES) {
        float d = deg[i];
        deg[i] = (d > 0.0f) ? rsqrtf(d) : 0.0f;
    }
}

__global__ void k_norm(const long long* __restrict__ row,
                       const long long* __restrict__ col,
                       const float* __restrict__ w,
                       const float* __restrict__ dinv,
                       float* __restrict__ norm) {
    int e = blockIdx.x * blockDim.x + threadIdx.x;
    if (e >= E_TOT) return;
    if (e < E_EDGES) {
        norm[e] = dinv[(int)row[e]] * w[e] * dinv[(int)col[e]];
    } else {
        float di = dinv[e - E_EDGES];
        norm[e] = di * di;                      // self loop: w=1, row==col
    }
}

// ---------------- bf16 conversion / packing ----------------
__global__ void k_cvt_x(const float* __restrict__ x, bf16* __restrict__ xbf) {
    int idx = blockIdx.x * blockDim.x + threadIdx.x;    // over N_NODES*K1PAD
    if (idx >= N_NODES * K1PAD) return;
    int i = idx >> 7;           // / 128
    int k = idx & (K1PAD - 1);
    xbf[idx] = (k < D_IN) ? (bf16)x[(size_t)i * D_IN + k] : (bf16)0.0f;
}

// W1 (D_IN x D_H row-major f32) -> W1t (D_H x K1PAD col-major-by-output bf16)
__global__ void k_cvt_w1(const float* __restrict__ W1, bf16* __restrict__ W1t) {
    int idx = blockIdx.x * blockDim.x + threadIdx.x;    // over D_H*K1PAD
    if (idx >= D_H * K1PAD) return;
    int n = idx >> 7;
    int k = idx & (K1PAD - 1);
    W1t[idx] = (k < D_IN) ? (bf16)W1[(size_t)k * D_H + n] : (bf16)0.0f;
}

// W2 (D_H x D_OUT row-major f32) -> W2t (D_OUT x D_H)
__global__ void k_cvt_w2(const float* __restrict__ W2, bf16* __restrict__ W2t) {
    int idx = blockIdx.x * blockDim.x + threadIdx.x;    // over D_OUT*D_H
    if (idx >= D_OUT * D_H) return;
    int n = idx >> 8;           // / 256
    int k = idx & (D_H - 1);
    W2t[idx] = (bf16)W2[(size_t)k * D_OUT + n];
}

// ---------------- WMMA GEMM: C[MxNc] = A[MxK](bf16 row-major) * Bt[NcxK](bf16) ----
// One wave computes a 16x64 tile: A-frag reused across 4 column tiles.
__global__ __launch_bounds__(256) void k_gemm_bf16(const bf16* __restrict__ A,
                                                   const bf16* __restrict__ Bt,
                                                   float* __restrict__ C,
                                                   int M, int Nc, int K) {
    int wave  = (blockIdx.x * blockDim.x + threadIdx.x) >> 5;
    int lane  = threadIdx.x & 31;
    int n64   = Nc >> 6;                 // number of 64-wide col tiles
    int mT    = wave / n64;
    int nBase = (wave % n64) << 6;
    if (mT >= (M >> 4)) return;          // wave-uniform: EXEC stays all-1s

    int half = lane >> 4;                // 0: lanes 0-15, 1: lanes 16-31
    int lsub = lane & 15;

    const bf16* arow = A + (size_t)((mT << 4) + lsub) * K;
    const bf16* bcol = Bt + (size_t)(nBase + lsub) * K;
    size_t bstep = (size_t)16 * K;       // next 16-column tile

    floatx8 acc0 = {}, acc1 = {}, acc2 = {}, acc3 = {};

    union AFrag { bf16x16 v; bf16x8 h[2]; };

    for (int kb = 0; kb < K; kb += 32) {
        // A 16x32 fragment, contiguous K to match B: lane<16 -> K {0..15}, lane>=16 -> {16..31}
        AFrag a;
        a.h[0] = *(const bf16x8*)(arow + kb + half * 16);
        a.h[1] = *(const bf16x8*)(arow + kb + half * 16 + 8);
        // B 32x16 fragments: lane's column, 16 contiguous K values (half selects K 0-15/16-31)
        int boff = kb + half * 16;
        bf16x16 b0 = *(const bf16x16*)(bcol + 0 * bstep + boff);
        bf16x16 b1 = *(const bf16x16*)(bcol + 1 * bstep + boff);
        bf16x16 b2 = *(const bf16x16*)(bcol + 2 * bstep + boff);
        bf16x16 b3 = *(const bf16x16*)(bcol + 3 * bstep + boff);
        acc0 = __builtin_amdgcn_wmma_f32_16x16x32_bf16(false, a.v, false, b0, (short)0, acc0, false, false);
        acc1 = __builtin_amdgcn_wmma_f32_16x16x32_bf16(false, a.v, false, b1, (short)0, acc1, false, false);
        acc2 = __builtin_amdgcn_wmma_f32_16x16x32_bf16(false, a.v, false, b2, (short)0, acc2, false, false);
        acc3 = __builtin_amdgcn_wmma_f32_16x16x32_bf16(false, a.v, false, b3, (short)0, acc3, false, false);
    }

    // C/D layout: VGPR r -> row (half*8 + r), col = lsub (per 16-wide tile)
    float* crow = C + (size_t)((mT << 4) + half * 8) * Nc + nBase + lsub;
#pragma unroll
    for (int r = 0; r < 8; ++r) {
        crow[(size_t)r * Nc + 0]  = acc0[r];
        crow[(size_t)r * Nc + 16] = acc1[r];
        crow[(size_t)r * Nc + 32] = acc2[r];
        crow[(size_t)r * Nc + 48] = acc3[r];
    }
}

// ---------------- aggregation (scatter-add), one block per edge ----------------
__global__ void k_scatter(const long long* __restrict__ row,
                          const long long* __restrict__ col,
                          const float* __restrict__ norm,
                          const float* __restrict__ H,
                          float* __restrict__ out, int D) {
    int e = blockIdx.x;                  // e in [0, E_TOT)
    int r, c;
    if (e < E_EDGES) { r = (int)row[e]; c = (int)col[e]; }
    else             { r = c = e - E_EDGES; }
    float nv = norm[e];
    int d = threadIdx.x;                 // blockDim.x == D
    atomicAdd(&out[(size_t)c * D + d], H[(size_t)r * D + d] * nv);
}

__global__ void k_zero4(float4* p, int n4) {
    int i = blockIdx.x * blockDim.x + threadIdx.x;
    if (i < n4) p[i] = make_float4(0.f, 0.f, 0.f, 0.f);
}

// h = relu(agg + b1) stored as bf16 for the second WMMA GEMM
__global__ void k_relu_bias_cvt(const float* __restrict__ agg,
                                const float* __restrict__ b1,
                                bf16* __restrict__ hbf) {
    int idx = blockIdx.x * blockDim.x + threadIdx.x;    // over N_NODES*D_H
    if (idx >= N_NODES * D_H) return;
    float v = agg[idx] + b1[idx & (D_H - 1)];
    hbf[idx] = (bf16)fmaxf(v, 0.0f);
}

// initialize output with bias b2 (scatter adds on top)
__global__ void k_out_init(float* __restrict__ out, const float* __restrict__ b2) {
    int idx = blockIdx.x * blockDim.x + threadIdx.x;    // over N_NODES*D_OUT
    if (idx < N_NODES * D_OUT) out[idx] = b2[idx & (D_OUT - 1)];
}

// ---------------- launcher ----------------
extern "C" void kernel_launch(void* const* d_in, const int* in_sizes, int n_in,
                              void* d_out, int out_size, void* d_ws, size_t ws_size,
                              hipStream_t stream) {
    const float*     x   = (const float*)d_in[0];
    const long long* ei  = (const long long*)d_in[1];   // int64 (2, E)
    const float*     ew  = (const float*)d_in[2];
    const float*     W1  = (const float*)d_in[3];
    const float*     b1  = (const float*)d_in[4];
    const float*     W2  = (const float*)d_in[5];
    const float*     b2  = (const float*)d_in[6];
    float* out = (float*)d_out;

    const long long* row = ei;            // sources
    const long long* col = ei + E_EDGES;  // targets

    // workspace carve-up (256B aligned)
    char* ws = (char*)d_ws;
    size_t off = 0;
    auto carve = [&](size_t bytes) -> char* {
        char* p = ws + off;
        off = (off + bytes + 255) & ~(size_t)255;
        return p;
    };
    float* deg  = (float*)carve((size_t)N_NODES * 4);          // becomes dinv in place
    float* norm = (float*)carve((size_t)E_TOT * 4);
    bf16*  xbf  = (bf16*) carve((size_t)N_NODES * K1PAD * 2);
    bf16*  W1t  = (bf16*) carve((size_t)D_H * K1PAD * 2);
    bf16*  W2t  = (bf16*) carve((size_t)D_OUT * D_H * 2);
    float* H1   = (float*)carve((size_t)N_NODES * D_H * 4);    // reused: hbf + H2
    float* AGG  = (float*)carve((size_t)N_NODES * D_H * 4);
    bf16*  hbf  = (bf16*)H1;                                   // first half of H1 region
    float* H2   = (float*)((char*)H1 + (size_t)N_NODES * D_H * 2); // second half

    const int T = 256;
    // ---- gcn_norm ----
    k_init_deg <<<(N_NODES + T - 1) / T, T, 0, stream>>>(deg);
    k_deg_edges<<<(E_EDGES + T - 1) / T, T, 0, stream>>>(col, ew, deg);
    k_dinv     <<<(N_NODES + T - 1) / T, T, 0, stream>>>(deg);
    k_norm     <<<(E_TOT + T - 1) / T, T, 0, stream>>>(row, col, ew, deg, norm);

    // ---- bf16 packing ----
    k_cvt_x <<<(N_NODES * K1PAD + T - 1) / T, T, 0, stream>>>(x, xbf);
    k_cvt_w1<<<(D_H * K1PAD + T - 1) / T, T, 0, stream>>>(W1, W1t);
    k_cvt_w2<<<(D_OUT * D_H + T - 1) / T, T, 0, stream>>>(W2, W2t);

    // ---- layer 1: H1 = x @ W1 (WMMA), aggregate, relu+bias ----
    {
        int waves = (N_NODES / 16) * (D_H / 64);   // 25000
        int blocks = (waves * 32 + T - 1) / T;     // 3125
        k_gemm_bf16<<<blocks, T, 0, stream>>>(xbf, W1t, H1, N_NODES, D_H, K1PAD);
    }
    k_zero4<<<((N_NODES * D_H / 4) + T - 1) / T, T, 0, stream>>>((float4*)AGG, N_NODES * D_H / 4);
    k_scatter<<<E_TOT, D_H, 0, stream>>>(row, col, norm, H1, AGG, D_H);
    k_relu_bias_cvt<<<(N_NODES * D_H + T - 1) / T, T, 0, stream>>>(AGG, b1, hbf);

    // ---- layer 2: H2 = h @ W2 (WMMA), aggregate into output (+b2) ----
    {
        int waves = (N_NODES / 16) * (D_OUT / 64); // 12500
        int blocks = (waves * 32 + T - 1) / T;     // 1563 (guard inside)
        k_gemm_bf16<<<blocks, T, 0, stream>>>(hbf, W2t, H2, N_NODES, D_OUT, D_H);
    }
    k_out_init<<<(N_NODES * D_OUT + T - 1) / T, T, 0, stream>>>(out, b2);
    k_scatter<<<E_TOT, D_OUT, 0, stream>>>(row, col, norm, H2, out, D_OUT);
}
